// SAGEVolModel_27195732918655
// MI455X (gfx1250) — matrix operand order repaired and checked by reading, not compiled, not run on hardware
//
#include <hip/hip_runtime.h>
#include <hip/hip_bf16.h>

typedef __attribute__((ext_vector_type(16))) __bf16 v16bf;
typedef __attribute__((ext_vector_type(8)))  float  v8f;

#define NN 100000
#define EE 1250000
#define DD 64
#define LSTR 68   // LDS row stride (floats) for the h buffer

// ---------------------------------------------------------------- utilities
__global__ __launch_bounds__(256) void zero_f32(float* __restrict__ p, int n) {
    int i = blockIdx.x * 256 + threadIdx.x;
    if (i < n) p[i] = 0.0f;
}

__global__ __launch_bounds__(256) void deg_kernel(const int* __restrict__ rows,
                                                  float* __restrict__ deg) {
    int e = blockIdx.x * 256 + threadIdx.x;
    if (e < EE) atomicAdd(&deg[rows[e]], 1.0f);
}

__global__ __launch_bounds__(256) void invdeg_kernel(const float* __restrict__ deg,
                                                     float* __restrict__ inv) {
    int i = blockIdx.x * 256 + threadIdx.x;
    if (i < NN) inv[i] = 1.0f / fmaxf(deg[i], 1.0f);
}

// gather x[col] -> atomic scatter-add into agg[row]; 16 threads/edge, 4 floats each
__global__ __launch_bounds__(256) void scatter_kernel(const int* __restrict__ rows,
                                                      const int* __restrict__ cols,
                                                      const float* __restrict__ x,
                                                      float* __restrict__ agg) {
    long long t = (long long)blockIdx.x * 256 + threadIdx.x;
    if (t >= (long long)EE * 16) return;
    int e = (int)(t >> 4);
    int c = ((int)t & 15) * 4;
    int dst = rows[e], src = cols[e];
    const float4 v = *(const float4*)(x + (long long)src * DD + c);
    float* a = agg + (long long)dst * DD + c;
    atomicAdd(a + 0, v.x);
    atomicAdd(a + 1, v.y);
    atomicAdd(a + 2, v.z);
    atomicAdd(a + 3, v.w);
}

// --------------------------------------------- weight -> B-fragment packing
// out layout: [tile(ncols/16)][kb(2)][lane(32)][e(16)] bf16
// ISA B layout: lane -> N = tile*16 + (lane&15); k = kb*32 + (e&7) + 8*(lane>>4) + 16*(e>>3)
__global__ __launch_bounds__(256) void pack_weights(const float* __restrict__ W,
                                                    __bf16* __restrict__ out,
                                                    int ncols) {
    int i = blockIdx.x * 256 + threadIdx.x;
    int total = (ncols / 16) * 2 * 32 * 16;
    if (i >= total) return;
    int e    = i & 15;
    int lane = (i >> 4) & 31;
    int kb   = (i >> 9) & 1;
    int tile = i >> 10;
    int k = kb * 32 + (e & 7) + 8 * (lane >> 4) + 16 * (e >> 3);
    int n = tile * 16 + (lane & 15);
    out[i] = (__bf16)W[k * ncols + n];
}

// A-fragment swizzle helper: float4 covering (m, kq..kq+3) lands at 4
// consecutive bf16 in fragment [kb][lane][e0..e0+3]:
//   kb   = kq>>5 ; kl = kq&31
//   half = (kl>>3)&1 ; e0 = (kl&7) | ((kl>>4)<<3) ; lane = m + 16*half
__device__ __forceinline__ int afrag_idx(int m, int kq) {
    int kb   = kq >> 5;
    int kl   = kq & 31;
    int half = (kl >> 3) & 1;
    int e0   = (kl & 7) | (((kl >> 4) & 1) << 3);
    int lane = m + 16 * half;
    return ((kb * 32 + lane) << 4) + e0;
}

// ------------------------------------------------- fused SAGE layer (WMMA)
// block = 128 threads = 4 waves; 16 rows per block; wave w -> cols [16w,16w+16)
__global__ __launch_bounds__(128)
void sage_layer_kernel(const float* __restrict__ x,
                       const float* __restrict__ agg,
                       const float* __restrict__ invdeg,
                       const __bf16* __restrict__ wsf,  // packed Ws frags [4][2][32][16]
                       const float* __restrict__ bsv,
                       const __bf16* __restrict__ wnf,  // packed Wn frags
                       const float* __restrict__ bnv,
                       const float* __restrict__ gamma,
                       const float* __restrict__ beta,
                       float* __restrict__ xout) {
    __shared__ __align__(32) __bf16 axf[2 * 32 * 16];  // x A-fragments (2 K-chunks)
    __shared__ __align__(32) __bf16 anf[2 * 32 * 16];  // neigh A-fragments
    __shared__ float hb[16 * LSTR];
    __shared__ float ps[16][8], ps2[16][8];
    __shared__ float mu_s[16], rs_s[16];

    const int tid  = threadIdx.x;
    const int lane = tid & 31;
    const int wave = tid >> 5;
    const int row0 = blockIdx.x * 16;

    // stage + convert + swizzle: 256 float4 tasks per matrix, 128 threads
#pragma unroll
    for (int it = 0; it < 2; ++it) {
        int i = tid + it * 128;
        int m = i >> 4, kq = (i & 15) * 4;
        long long g = ((long long)(row0 + m) * DD + kq) >> 2;
        float4 vx = ((const float4*)x)[g];
        float4 va = ((const float4*)agg)[g];
        float s = invdeg[row0 + m];
        int d = afrag_idx(m, kq);
        axf[d + 0] = (__bf16)vx.x; axf[d + 1] = (__bf16)vx.y;
        axf[d + 2] = (__bf16)vx.z; axf[d + 3] = (__bf16)vx.w;
        anf[d + 0] = (__bf16)(va.x * s); anf[d + 1] = (__bf16)(va.y * s);
        anf[d + 2] = (__bf16)(va.z * s); anf[d + 3] = (__bf16)(va.w * s);
    }
    __syncthreads();

    // operand fragments: contiguous 32B per lane
    const v16bf* axp = (const v16bf*)axf;
    const v16bf* anp = (const v16bf*)anf;
    v16bf ax0 = axp[lane],      an0 = anp[lane];
    v16bf ax1 = axp[32 + lane], an1 = anp[32 + lane];

    const v16bf* wsp = (const v16bf*)wsf;  // [tile*2+kb][lane]
    const v16bf* wnp = (const v16bf*)wnf;
    v16bf wsb0 = wsp[(wave * 2 + 0) * 32 + lane];
    v16bf wsb1 = wsp[(wave * 2 + 1) * 32 + lane];
    v16bf wnb0 = wnp[(wave * 2 + 0) * 32 + lane];
    v16bf wnb1 = wnp[(wave * 2 + 1) * 32 + lane];

    v8f c = {};
    c = __builtin_amdgcn_wmma_f32_16x16x32_bf16(false, ax0, false, wsb0, (short)0, c, false, false);
    c = __builtin_amdgcn_wmma_f32_16x16x32_bf16(false, ax1, false, wsb1, (short)0, c, false, false);
    c = __builtin_amdgcn_wmma_f32_16x16x32_bf16(false, an0, false, wnb0, (short)0, c, false, false);
    c = __builtin_amdgcn_wmma_f32_16x16x32_bf16(false, an1, false, wnb1, (short)0, c, false, false);

    // bias + relu; C/D layout: lane -> N = wave*16 + (lane&15), VGPR r -> M = r + 8*(lane>>4)
    const int half = lane >> 4;
    const int l16  = lane & 15;
    const int n0   = wave * 16;
    float bsum = bsv[n0 + l16] + bnv[n0 + l16];
#pragma unroll
    for (int r = 0; r < 8; ++r) {
        float v = fmaxf(c[r] + bsum, 0.0f);
        hb[(r + 8 * half) * LSTR + n0 + l16] = v;
    }
    __syncthreads();

    // layernorm stats: 128 threads -> 8-element partials, then 16 threads finish
    {
        int m = tid >> 3, seg = tid & 7;
        float s = 0.0f, s2 = 0.0f;
#pragma unroll
        for (int j = 0; j < 8; ++j) {
            float v = hb[m * LSTR + seg * 8 + j];
            s += v; s2 += v * v;
        }
        ps[m][seg] = s; ps2[m][seg] = s2;
    }
    __syncthreads();
    if (tid < 16) {
        float s = 0.0f, s2 = 0.0f;
#pragma unroll
        for (int j = 0; j < 8; ++j) { s += ps[tid][j]; s2 += ps2[tid][j]; }
        float mu  = s * (1.0f / DD);
        float var = s2 * (1.0f / DD) - mu * mu;
        mu_s[tid] = mu;
        rs_s[tid] = rsqrtf(var + 1e-5f);
    }
    __syncthreads();

    // normalize + affine, vectorized stores
#pragma unroll
    for (int it = 0; it < 2; ++it) {
        int i = tid + it * 128;
        int m = i >> 4, kq = (i & 15) * 4;
        float mu = mu_s[m], rs = rs_s[m];
        float4 o;
        o.x = (hb[m * LSTR + kq + 0] - mu) * rs * gamma[kq + 0] + beta[kq + 0];
        o.y = (hb[m * LSTR + kq + 1] - mu) * rs * gamma[kq + 1] + beta[kq + 1];
        o.z = (hb[m * LSTR + kq + 2] - mu) * rs * gamma[kq + 2] + beta[kq + 2];
        o.w = (hb[m * LSTR + kq + 3] - mu) * rs * gamma[kq + 3] + beta[kq + 3];
        ((float4*)xout)[((long long)(row0 + m) * DD + kq) >> 2] = o;
    }
}

// ------------------------------------------------------------- head (WMMA)
// block = 64 threads = 2 waves; 16 rows; h1 = relu(x@hw1+hb1) (16x32), out = h1@hw2+hb2
__global__ __launch_bounds__(64)
void head_kernel(const float* __restrict__ x,
                 const __bf16* __restrict__ hw1f,  // packed [2][2][32][16]
                 const float* __restrict__ hb1,
                 const float* __restrict__ hw2,
                 const float* __restrict__ hb2,
                 float* __restrict__ out) {
    __shared__ __align__(32) __bf16 axf[2 * 32 * 16];
    __shared__ float h1[16 * 36];

    const int tid  = threadIdx.x;
    const int lane = tid & 31;
    const int wave = tid >> 5;
    const int row0 = blockIdx.x * 16;

#pragma unroll
    for (int it = 0; it < 4; ++it) {
        int i = tid + it * 64;
        int m = i >> 4, kq = (i & 15) * 4;
        float4 vx = ((const float4*)x)[((long long)(row0 + m) * DD + kq) >> 2];
        int d = afrag_idx(m, kq);
        axf[d + 0] = (__bf16)vx.x; axf[d + 1] = (__bf16)vx.y;
        axf[d + 2] = (__bf16)vx.z; axf[d + 3] = (__bf16)vx.w;
    }
    __syncthreads();

    const v16bf* axp = (const v16bf*)axf;
    v16bf a0 = axp[lane], a1 = axp[32 + lane];
    const v16bf* wp = (const v16bf*)hw1f;
    v16bf b0 = wp[(wave * 2 + 0) * 32 + lane];
    v16bf b1 = wp[(wave * 2 + 1) * 32 + lane];

    v8f c = {};
    c = __builtin_amdgcn_wmma_f32_16x16x32_bf16(false, a0, false, b0, (short)0, c, false, false);
    c = __builtin_amdgcn_wmma_f32_16x16x32_bf16(false, a1, false, b1, (short)0, c, false, false);

    const int half = lane >> 4;
    const int l16  = lane & 15;
    const int n0   = wave * 16;
    float bias = hb1[n0 + l16];
#pragma unroll
    for (int r = 0; r < 8; ++r)
        h1[(r + 8 * half) * 36 + n0 + l16] = fmaxf(c[r] + bias, 0.0f);
    __syncthreads();

    if (tid < 16) {
        float s = hb2[0];
#pragma unroll
        for (int j = 0; j < 32; ++j) s += h1[tid * 36 + j] * hw2[j];
        out[row0 + tid] = s;
    }
}

// ---------------------------------------------------------------- launcher
extern "C" void kernel_launch(void* const* d_in, const int* in_sizes, int n_in,
                              void* d_out, int out_size, void* d_ws, size_t ws_size,
                              hipStream_t stream) {
    const float* x0    = (const float*)d_in[0];
    const int*   ei    = (const int*)d_in[1];
    const float* Ws    = (const float*)d_in[2];
    const float* bs    = (const float*)d_in[3];
    const float* Wn    = (const float*)d_in[4];
    const float* bn    = (const float*)d_in[5];
    const float* gamma = (const float*)d_in[6];
    const float* beta  = (const float*)d_in[7];
    const float* hw1   = (const float*)d_in[8];
    const float* hb1   = (const float*)d_in[9];
    const float* hw2   = (const float*)d_in[10];
    const float* hb2   = (const float*)d_in[11];

    const int* rows = ei;        // edge_index[0] = dst
    const int* cols = ei + EE;   // edge_index[1] = src

    float* bufA = (float*)d_ws;              // N*D
    float* bufB = bufA + (size_t)NN * DD;    // N*D
    float* agg  = bufB + (size_t)NN * DD;    // N*D
    float* deg  = agg  + (size_t)NN * DD;    // N
    float* invd = deg  + NN;                 // N
    // bf16 fragment region (32B-aligned: 19,400,000 floats -> 77,600,000 B)
    __bf16* wsf  = (__bf16*)(invd + NN);     // 3 * 4096
    __bf16* wnf  = wsf + 3 * 4096;           // 3 * 4096
    __bf16* h1f  = wnf + 3 * 4096;           // 2048

    const int nThreads = 256;
    const int ND = NN * DD;

    // pack weights into WMMA B-fragment order (bf16)
    for (int l = 0; l < 3; ++l) {
        pack_weights<<<(4096 + 255) / 256, 256, 0, stream>>>(Ws + (size_t)l * DD * DD, wsf + l * 4096, DD);
        pack_weights<<<(4096 + 255) / 256, 256, 0, stream>>>(Wn + (size_t)l * DD * DD, wnf + l * 4096, DD);
    }
    pack_weights<<<(2048 + 255) / 256, 256, 0, stream>>>(hw1, h1f, 32);

    // degree (per call, replay-safe)
    zero_f32<<<(NN + nThreads - 1) / nThreads, nThreads, 0, stream>>>(deg, NN);
    deg_kernel<<<(EE + nThreads - 1) / nThreads, nThreads, 0, stream>>>(rows, deg);
    invdeg_kernel<<<(NN + nThreads - 1) / nThreads, nThreads, 0, stream>>>(deg, invd);

    const float* xin = x0;
    float* xout = bufA;
    const long long scatterThreads = (long long)EE * 16;
    const int scatterBlocks = (int)((scatterThreads + nThreads - 1) / nThreads);

    for (int l = 0; l < 3; ++l) {
        zero_f32<<<(ND + nThreads - 1) / nThreads, nThreads, 0, stream>>>(agg, ND);
        scatter_kernel<<<scatterBlocks, nThreads, 0, stream>>>(rows, cols, xin, agg);
        sage_layer_kernel<<<NN / 16, 128, 0, stream>>>(
            xin, agg, invd,
            wsf + l * 4096, bs + l * DD,
            wnf + l * 4096, bn + l * DD,
            gamma + l * DD, beta + l * DD, xout);
        xin  = xout;
        xout = (l == 0) ? bufB : bufA;
    }

    head_kernel<<<NN / 16, 64, 0, stream>>>(xin, h1f, hb1, hw2, hb2, (float*)d_out);
}
